// HierarchicalLSTM_6004364280380
// MI455X (gfx1250) — matrix operand order repaired and checked by reading, not compiled, not run on hardware
//
#include <hip/hip_runtime.h>
#include <hip/hip_bf16.h>
#include <stdint.h>
#include <stddef.h>

// ---------------- problem constants ----------------
#define T_    80
#define B_    512
#define HDIM_ 1024
#define H_    512
#define S_    13
#define A_    121
#define AP_   128          // A padded to multiple of 16
#define P_    256
#define TB_   (T_ * B_)    // 40960
#define G4H_  (4 * H_)     // 2048

// Async global->LDS path (CDNA5); guarded so absence falls back cleanly.
#if defined(__AMDGCN__) && __has_builtin(__builtin_amdgcn_global_load_async_to_lds_b128)
#define HAVE_ASYNC_LDS 1
#endif

// ---------------- WMMA types ----------------
typedef __attribute__((ext_vector_type(8)))  float          v8f;
typedef __attribute__((ext_vector_type(16))) __bf16         bf16x16;
typedef __attribute__((ext_vector_type(8)))  unsigned short u16x8;
typedef __attribute__((ext_vector_type(4)))  int            v4i;

#ifdef HAVE_ASYNC_LDS
typedef __attribute__((address_space(1))) v4i* gptr_v4i;   // global int4*
typedef __attribute__((address_space(3))) v4i* lptr_v4i;   // LDS int4*
#endif

union Frag {
    u16x8   h[2];
    bf16x16 v;
};

__device__ __forceinline__ v8f vzero8() {
    v8f v;
#pragma unroll
    for (int i = 0; i < 8; ++i) v[i] = 0.0f;
    return v;
}

__device__ __forceinline__ unsigned short f2bf(float f) {
    unsigned int x = __float_as_uint(f);
    unsigned int r = x + 0x7fffu + ((x >> 16) & 1u);   // round-to-nearest-even
    return (unsigned short)(r >> 16);
}

__device__ __forceinline__ float sigm(float x) {
    return 1.0f / (1.0f + __expf(-x));
}

// A fragment (16x32 bf16): lane L holds row M = L%16.
// element e -> k = (e/8)*16 + (L/16)*8 + e%8  => two contiguous 16B runs.
__device__ __forceinline__ Frag load_a_frag(const unsigned short* base, int lda,
                                            int row, int k0, int lane) {
    Frag f;
    const unsigned short* p = base + (size_t)row * lda + k0 + ((lane >> 4) << 3);
    f.h[0] = *(const u16x8*)(p);
    f.h[1] = *(const u16x8*)(p + 16);
    return f;
}

// B fragment (32x16 bf16, sourced from row-major W[N][K], B[k][n]=W[n][k]):
// lane L holds col N = L%16; element e -> k = (e/8)*16 + (L/16)*8 + e%8.
__device__ __forceinline__ Frag load_b_frag(const unsigned short* base, int ldb,
                                            int col, int k0, int lane) {
    Frag f;
    const unsigned short* p = base + (size_t)col * ldb + k0 + ((lane >> 4) << 3);
    f.h[0] = *(const u16x8*)(p);
    f.h[1] = *(const u16x8*)(p + 16);
    return f;
}

__device__ __forceinline__ v8f wmma_bf16(const Frag& a, const Frag& b, v8f c) {
    return __builtin_amdgcn_wmma_f32_16x16x32_bf16(
        /*neg_a=*/false, a.v, /*neg_b=*/false, b.v,
        /*c_mod=*/(short)0, c, /*reuse_a=*/false, /*reuse_b=*/false);
}

// ---------------- conversion kernels ----------------
__global__ void cvt_f32_bf16(const float* __restrict__ src,
                             unsigned short* __restrict__ dst, size_t n) {
    size_t i = (size_t)blockIdx.x * blockDim.x + threadIdx.x;
    size_t stride = (size_t)gridDim.x * blockDim.x;
    for (; i < n; i += stride) dst[i] = f2bf(src[i]);
}

// Wp2 [S,A,P] -> padded bf16 [S,AP,P] (rows 121..127 zero)
__global__ void cvt_wp2_pad(const float* __restrict__ src,
                            unsigned short* __restrict__ dst) {
    int i = blockIdx.x * blockDim.x + threadIdx.x;
    const int total = S_ * AP_ * P_;
    if (i >= total) return;
    int k = i & (P_ - 1);
    int n = (i >> 8) & (AP_ - 1);
    int s = i >> 15;
    dst[i] = (n < A_) ? f2bf(src[((size_t)s * A_ + n) * P_ + k]) : (unsigned short)0;
}

__global__ void bias_sum(const float* __restrict__ b_ih,
                         const float* __restrict__ b_hh,
                         float* __restrict__ bsum) {
    int i = blockIdx.x * blockDim.x + threadIdx.x;
    if (i < G4H_) bsum[i] = b_ih[i] + b_hh[i];
}

// masked h -> bf16 (episode-boundary mask applied to the recurrent input)
__global__ void mask_cvt_h(const float* __restrict__ h,
                           const unsigned char* __restrict__ done,
                           unsigned short* __restrict__ hmb, int t) {
    int i = blockIdx.x * blockDim.x + threadIdx.x;
    if (i >= B_ * H_) return;
    int b = i >> 9;
    float m = done[(size_t)t * B_ + b] ? 0.0f : 1.0f;
    hmb[i] = f2bf(h[i] * m);
}

// ---------------- LSTM gates GEMM (fused x@Wih^T + hm@Whh^T + bias) ----------
// gates[B_, 4H] ; each wave computes a 16x64 tile; grid = 1024 waves.
__global__ void lstm_gates_kernel(const unsigned short* __restrict__ Xb,   // [T*B, HDIM]
                                  const unsigned short* __restrict__ hmb,  // [B, H]
                                  const unsigned short* __restrict__ Wih,  // [4H, HDIM]
                                  const unsigned short* __restrict__ Whh,  // [4H, H]
                                  const float* __restrict__ bsum,          // [4H]
                                  float* __restrict__ gates,               // [B, 4H]
                                  int t) {
    const int lane = threadIdx.x & 31;
    const int wid  = threadIdx.x >> 5;
    const int gw   = blockIdx.x * (blockDim.x >> 5) + wid;
    const int tilesN = G4H_ / 64;               // 32
    const int tm = gw / tilesN;                 // 0..31
    const int tn = gw % tilesN;
    if (tm >= B_ / 16) return;

    v8f acc[4];
#pragma unroll
    for (int j = 0; j < 4; ++j) acc[j] = vzero8();

    const int row = tm * 16 + (lane & 15);
    const unsigned short* Xrow = Xb + (size_t)t * B_ * HDIM_;

    // K part 1: x @ Wih^T  (K = 1024)
    for (int k0 = 0; k0 < HDIM_; k0 += 32) {
        Frag a = load_a_frag(Xrow, HDIM_, row, k0, lane);
#pragma unroll
        for (int j = 0; j < 4; ++j) {
            int col = tn * 64 + j * 16 + (lane & 15);
            Frag b = load_b_frag(Wih, HDIM_, col, k0, lane);
            acc[j] = wmma_bf16(a, b, acc[j]);
        }
    }
    // K part 2: hm @ Whh^T  (K = 512)
    for (int k0 = 0; k0 < H_; k0 += 32) {
        Frag a = load_a_frag(hmb, H_, row, k0, lane);
#pragma unroll
        for (int j = 0; j < 4; ++j) {
            int col = tn * 64 + j * 16 + (lane & 15);
            Frag b = load_b_frag(Whh, H_, col, k0, lane);
            acc[j] = wmma_bf16(a, b, acc[j]);
        }
    }
    // epilogue: D layout: vgpr r -> M = r + 8*(lane/16), N = lane%16
    const int half = lane >> 4;
#pragma unroll
    for (int j = 0; j < 4; ++j) {
        int n = tn * 64 + j * 16 + (lane & 15);
        float bn = bsum[n];
#pragma unroll
        for (int r = 0; r < 8; ++r) {
            int m = tm * 16 + r + half * 8;
            gates[(size_t)m * G4H_ + n] = acc[j][r] + bn;
        }
    }
}

// ---------------- LSTM pointwise ----------------
__global__ void lstm_pointwise(const float* __restrict__ gates,
                               const unsigned char* __restrict__ done,
                               float* __restrict__ h, float* __restrict__ c,
                               float* __restrict__ coreF,
                               unsigned short* __restrict__ coreB, int t) {
    int i = blockIdx.x * blockDim.x + threadIdx.x;
    if (i >= B_ * H_) return;
    int b  = i >> 9;
    int hh = i & (H_ - 1);
    float m = done[(size_t)t * B_ + b] ? 0.0f : 1.0f;
    const float* g = gates + (size_t)b * G4H_;
    float ig = g[hh];
    float fg = g[H_ + hh];
    float gg = g[2 * H_ + hh];
    float og = g[3 * H_ + hh];
    float cold = c[i] * m;
    float cn = sigm(fg) * cold + sigm(ig) * tanhf(gg);
    float hn = sigm(og) * tanhf(cn);
    c[i] = cn;
    h[i] = hn;
    coreF[(size_t)t * B_ * H_ + i] = hn;
    coreB[(size_t)t * B_ * H_ + i] = f2bf(hn);
}

// ---------------- fused policy heads (Linear -> ELU -> Linear) ---------------
// grid: (TB/64, S). Block 256 threads = 8 waves.
// The 64x512 bf16 activation block is staged once into LDS (async copy on
// CDNA5), stage-1/2 A fragments come from LDS, h1 lives only in LDS.
__global__ void policy_kernel(const unsigned short* __restrict__ coreB, // [TB, H]
                              const unsigned short* __restrict__ Wp1b,  // [S, P, H]
                              const float* __restrict__ bp1,            // [S, P]
                              const unsigned short* __restrict__ Wp2b,  // [S, AP, P]
                              const float* __restrict__ bp2,            // [S, A]
                              float* __restrict__ out2)                 // [TB, A, S]
{
    __shared__ __align__(16) unsigned short asmem[64][H_ + 8];  // 64x520 ush = 66,560 B
    __shared__ __align__(16) unsigned short h1s[64][P_ + 8];    // 64x264 ush = 33,792 B
    const int s = blockIdx.y;
    const int rowbase = blockIdx.x * 64;
    const int lane = threadIdx.x & 31;
    const int wid  = threadIdx.x >> 5;      // 0..7
    const int half = lane >> 4;

    // ---- stage 0: stage A block [64 x 512] bf16 into LDS (64 KB) ----
    {
        const unsigned short* gsrc = coreB + (size_t)rowbase * H_;
        // 4096 chunks of 16B; 256 threads -> 16 iterations
#pragma unroll
        for (int it = 0; it < 16; ++it) {
            int idx = it * 256 + (int)threadIdx.x;     // 16B-chunk index
            int row = idx >> 6;                        // 64 chunks per row
            int col = (idx & 63) << 3;                 // ushort column
            const unsigned short* g = gsrc + (size_t)row * H_ + col;
#ifdef HAVE_ASYNC_LDS
            __builtin_amdgcn_global_load_async_to_lds_b128(
                (gptr_v4i)(const void*)g,
                (lptr_v4i)(void*)&asmem[row][col],
                0, 0);
#else
            *(u16x8*)&asmem[row][col] = *(const u16x8*)g;
#endif
        }
#ifdef HAVE_ASYNC_LDS
#if __has_builtin(__builtin_amdgcn_s_wait_asynccnt)
        __builtin_amdgcn_s_wait_asynccnt(0);
#else
        asm volatile("s_wait_asynccnt 0" ::: "memory");
#endif
#endif
    }
    __syncthreads();

    // ---- stage 1: H1[64,256] = elu(core @ Wp1[s]^T + bp1[s]) -> LDS bf16 ----
    {
        const unsigned short* W1 = Wp1b + (size_t)s * P_ * H_;
        const int mt = wid & 3;             // m-tile 0..3
        const int ng = (wid >> 2) * 8;      // n-tile base: 0 or 8
        v8f acc[8];
#pragma unroll
        for (int j = 0; j < 8; ++j) acc[j] = vzero8();
        const int mrow = mt * 16 + (lane & 15);
        for (int k0 = 0; k0 < H_; k0 += 32) {
            Frag a;                         // A from LDS: two ds_load_b128
            const unsigned short* p = &asmem[mrow][k0 + (half << 3)];
            a.h[0] = *(const u16x8*)(p);
            a.h[1] = *(const u16x8*)(p + 16);
#pragma unroll
            for (int j = 0; j < 8; ++j) {
                int col = (ng + j) * 16 + (lane & 15);
                Frag b = load_b_frag(W1, H_, col, k0, lane);
                acc[j] = wmma_bf16(a, b, acc[j]);
            }
        }
#pragma unroll
        for (int j = 0; j < 8; ++j) {
            int n = (ng + j) * 16 + (lane & 15);
            float bias = bp1[s * P_ + n];
#pragma unroll
            for (int r = 0; r < 8; ++r) {
                int mr = mt * 16 + r + half * 8;
                float v = acc[j][r] + bias;
                v = (v > 0.0f) ? v : (__expf(v) - 1.0f);   // ELU (alpha=1)
                h1s[mr][n] = f2bf(v);
            }
        }
    }
    __syncthreads();

    // ---- stage 2: logits[64,128(121)] = H1 @ Wp2[s]^T + bp2[s] ----
    {
        const unsigned short* W2 = Wp2b + (size_t)s * AP_ * P_;
        const int mt  = wid & 3;
        const int nt0 = (wid >> 2) * 4;     // n-tile base: 0 or 4
        v8f acc[4];
#pragma unroll
        for (int j = 0; j < 4; ++j) acc[j] = vzero8();
        const int mrow = mt * 16 + (lane & 15);
        for (int k0 = 0; k0 < P_; k0 += 32) {
            Frag a;                         // A from LDS: two ds_load_b128
            const unsigned short* p = &h1s[mrow][k0 + (half << 3)];
            a.h[0] = *(const u16x8*)(p);
            a.h[1] = *(const u16x8*)(p + 16);
#pragma unroll
            for (int j = 0; j < 4; ++j) {
                int col = (nt0 + j) * 16 + (lane & 15);
                Frag b = load_b_frag(W2, P_, col, k0, lane);
                acc[j] = wmma_bf16(a, b, acc[j]);
            }
        }
#pragma unroll
        for (int j = 0; j < 4; ++j) {
            int n = (nt0 + j) * 16 + (lane & 15);
            if (n < A_) {
                float bias = bp2[s * A_ + n];
#pragma unroll
                for (int r = 0; r < 8; ++r) {
                    int row = rowbase + mt * 16 + r + half * 8;
                    out2[((size_t)row * A_ + n) * S_ + s] = acc[j][r] + bias;
                }
            }
        }
    }
}

// ---------------- strategy logits + gumbel argmax + baseline -----------------
__global__ void strat_kernel(const float* __restrict__ coreF,   // [TB, H]
                             const float* __restrict__ Ws, const float* __restrict__ bs,
                             const float* __restrict__ Wb, const float* __restrict__ bb,
                             const float* __restrict__ gu,      // [TB, S]
                             float* __restrict__ out6,          // strategy logits [TB,S]
                             float* __restrict__ out3,          // baseline [TB]
                             int* __restrict__ sidx)            // [TB]
{
    const int lane = threadIdx.x & 31;
    const int row  = blockIdx.x * (blockDim.x >> 5) + (threadIdx.x >> 5);
    if (row >= TB_) return;
    float acc[S_];
#pragma unroll
    for (int j = 0; j < S_; ++j) acc[j] = 0.0f;
    float accb = 0.0f;
    const float* x = coreF + (size_t)row * H_;
    for (int k = lane; k < H_; k += 32) {
        float xv = x[k];
#pragma unroll
        for (int j = 0; j < S_; ++j) acc[j] += xv * Ws[j * H_ + k];
        accb += xv * Wb[k];
    }
#pragma unroll
    for (int j = 0; j < S_; ++j)
        for (int off = 16; off > 0; off >>= 1) acc[j] += __shfl_xor(acc[j], off, 32);
    for (int off = 16; off > 0; off >>= 1) accb += __shfl_xor(accb, off, 32);

    if (lane == 0) {
        float best = -3.4e38f;
        int bi = 0;
#pragma unroll
        for (int j = 0; j < S_; ++j) {
            float lg = acc[j] + bs[j];
            out6[(size_t)row * S_ + j] = lg;
            float u = gu[(size_t)row * S_ + j];
            float g = -__logf(-__logf(u + 1e-10f) + 1e-10f);
            float sc = lg + g;                       // TAU == 1
            if (sc > best) { best = sc; bi = j; }
        }
        sidx[row] = bi;
        out3[row] = accb + bb[0];
    }
}

// ---------------- strategy selection + action sampling -----------------------
__global__ void action_kernel(const float* __restrict__ out2,   // [TB, A, S]
                              const int* __restrict__ sidx,
                              const float* __restrict__ au,     // [TB, A]
                              float* __restrict__ out1,         // [TB, A]
                              float* __restrict__ out4,         // action [TB]
                              float* __restrict__ out5)         // version [TB]
{
    const int lane = threadIdx.x & 31;
    const int row  = blockIdx.x * (blockDim.x >> 5) + (threadIdx.x >> 5);
    if (row >= TB_) return;
    const int s = sidx[row];
    float bestk = -3.4e38f;
    int bestn = 0;
    for (int n = lane; n < A_; n += 32) {
        float v = out2[((size_t)row * A_ + n) * S_ + s];
        out1[(size_t)row * A_ + n] = v;
        float u = au[(size_t)row * A_ + n];
        float g = -__logf(-__logf(u + 1e-10f) + 1e-10f);
        float key = v + g;
        if (key > bestk) { bestk = key; bestn = n; }
    }
    for (int off = 16; off > 0; off >>= 1) {
        float ok = __shfl_xor(bestk, off, 32);
        int   on = __shfl_xor(bestn, off, 32);
        if (ok > bestk || (ok == bestk && on < bestn)) { bestk = ok; bestn = on; }
    }
    if (lane == 0) {
        out4[row] = (float)bestn;
        out5[row] = 0.0f;
    }
}

// ---------------- host-side launch ----------------
extern "C" void kernel_launch(void* const* d_in, const int* in_sizes, int n_in,
                              void* d_out, int out_size, void* d_ws, size_t ws_size,
                              hipStream_t stream) {
    (void)in_sizes; (void)n_in; (void)out_size; (void)ws_size;

    const float*         core_input = (const float*)d_in[0];
    const unsigned char* done       = (const unsigned char*)d_in[1];
    const float*         h0         = (const float*)d_in[2];
    const float*         c0         = (const float*)d_in[3];
    const float*         W_ih       = (const float*)d_in[4];
    const float*         W_hh       = (const float*)d_in[5];
    const float*         b_ih       = (const float*)d_in[6];
    const float*         b_hh       = (const float*)d_in[7];
    const float*         Ws         = (const float*)d_in[8];
    const float*         bs         = (const float*)d_in[9];
    const float*         Wp1        = (const float*)d_in[10];
    const float*         bp1        = (const float*)d_in[11];
    const float*         Wp2        = (const float*)d_in[12];
    const float*         bp2        = (const float*)d_in[13];
    const float*         Wb         = (const float*)d_in[14];
    const float*         bb         = (const float*)d_in[15];
    const float*         gu         = (const float*)d_in[16];
    const float*         au         = (const float*)d_in[17];

    // ---- workspace layout (bytes) ----
    char* ws = (char*)d_ws;
    size_t off = 0;
    unsigned short* Xb    = (unsigned short*)(ws + off); off += (size_t)TB_ * HDIM_ * 2;   // 83.9 MB
    unsigned short* Wihb  = (unsigned short*)(ws + off); off += (size_t)G4H_ * HDIM_ * 2;  // 4.2 MB
    unsigned short* Whhb  = (unsigned short*)(ws + off); off += (size_t)G4H_ * H_ * 2;     // 2.1 MB
    unsigned short* Wp1b  = (unsigned short*)(ws + off); off += (size_t)S_ * P_ * H_ * 2;  // 3.4 MB
    unsigned short* Wp2b  = (unsigned short*)(ws + off); off += (size_t)S_ * AP_ * P_ * 2; // 0.85 MB
    float*          coreF = (float*)(ws + off);          off += (size_t)TB_ * H_ * 4;      // 83.9 MB
    unsigned short* coreB = (unsigned short*)(ws + off); off += (size_t)TB_ * H_ * 2;      // 41.9 MB
    float*          hbuf  = (float*)(ws + off);          off += (size_t)B_ * H_ * 4;
    float*          cbuf  = (float*)(ws + off);          off += (size_t)B_ * H_ * 4;
    unsigned short* hmb   = (unsigned short*)(ws + off); off += (size_t)B_ * H_ * 2;
    float*          gates = (float*)(ws + off);          off += (size_t)B_ * G4H_ * 4;
    float*          bsum  = (float*)(ws + off);          off += (size_t)G4H_ * 4;
    int*            sidx  = (int*)(ws + off);            off += (size_t)TB_ * 4;

    // ---- output layout (flat floats, reference return order) ----
    float* out = (float*)d_out;
    float* out1 = out;                                         // [T,B,A]
    float* out2 = out1 + (size_t)TB_ * A_;                     // [TB,A,S]
    float* out3 = out2 + (size_t)TB_ * A_ * S_;                // baseline [T,B]
    float* out4 = out3 + (size_t)TB_;                          // action [T,B]
    float* out5 = out4 + (size_t)TB_;                          // version [T,B]
    float* out6 = out5 + (size_t)TB_;                          // strategy logits [T,B,S]
    float* out7 = out6 + (size_t)TB_ * S_;                     // hT [1,B,H]
    float* out8 = out7 + (size_t)B_ * H_;                      // cT [1,B,H]

    // ---- precision conversion (one-time per call) ----
    cvt_f32_bf16<<<4096, 256, 0, stream>>>(core_input, Xb,   (size_t)TB_ * HDIM_);
    cvt_f32_bf16<<<2048, 256, 0, stream>>>(W_ih,       Wihb, (size_t)G4H_ * HDIM_);
    cvt_f32_bf16<<<1024, 256, 0, stream>>>(W_hh,       Whhb, (size_t)G4H_ * H_);
    cvt_f32_bf16<<<1024, 256, 0, stream>>>(Wp1,        Wp1b, (size_t)S_ * P_ * H_);
    cvt_wp2_pad <<<(S_ * AP_ * P_ + 255) / 256, 256, 0, stream>>>(Wp2, Wp2b);
    bias_sum    <<<(G4H_ + 255) / 256, 256, 0, stream>>>(b_ih, b_hh, bsum);

    (void)hipMemcpyAsync(hbuf, h0, (size_t)B_ * H_ * 4, hipMemcpyDeviceToDevice, stream);
    (void)hipMemcpyAsync(cbuf, c0, (size_t)B_ * H_ * 4, hipMemcpyDeviceToDevice, stream);

    // ---- sequential LSTM over T steps ----
    const int gatesWaves  = (B_ / 16) * (G4H_ / 64);  // 1024 waves
    const int gatesBlocks = gatesWaves / 4;           // 128 thr = 4 waves/block
    for (int t = 0; t < T_; ++t) {
        mask_cvt_h      <<<(B_ * H_ + 255) / 256, 256, 0, stream>>>(hbuf, done, hmb, t);
        lstm_gates_kernel<<<gatesBlocks, 128, 0, stream>>>(Xb, hmb, Wihb, Whhb, bsum, gates, t);
        lstm_pointwise  <<<(B_ * H_ + 255) / 256, 256, 0, stream>>>(gates, done, hbuf, cbuf,
                                                                    coreF, coreB, t);
    }

    // ---- all policy heads (fused 2-layer MLP, LDS-staged activations) ----
    policy_kernel<<<dim3(TB_ / 64, S_), 256, 0, stream>>>(coreB, Wp1b, bp1, Wp2b, bp2, out2);

    // ---- strategy logits + gumbel argmax + baseline ----
    strat_kernel<<<TB_ / 8, 256, 0, stream>>>(coreF, Ws, bs, Wb, bb, gu, out6, out3, sidx);

    // ---- select strategy logits + sample action ----
    action_kernel<<<TB_ / 8, 256, 0, stream>>>(out2, sidx, au, out1, out4, out5);

    // ---- final LSTM state ----
    (void)hipMemcpyAsync(out7, hbuf, (size_t)B_ * H_ * 4, hipMemcpyDeviceToDevice, stream);
    (void)hipMemcpyAsync(out8, cbuf, (size_t)B_ * H_ * 4, hipMemcpyDeviceToDevice, stream);
}
